// PINodeSTGNN_23364622090375
// MI455X (gfx1250) — matrix operand order repaired and checked by reading, not compiled, not run on hardware
//
#include <hip/hip_runtime.h>
#include <hip/hip_bf16.h>
#include <cmath>

// ---------------------------------------------------------------------------
// PINodeSTGNN for MI455X (gfx1250, wave32, WMMA)
//  - GAT projections + ODE MLP GEMMs on v_wmma_f32_16x16x32_f16 (f32 accum)
//  - WMMA B-operands (weights) pre-packed ONCE into fragment-major f16 images
//    in workspace; per-block LDS fill is a straight b128 copy
//  - WMMA A-operands staged in LDS fragment-major: reads = 2x ds_load_b128
//  - edge softmax/scatter via f32 global atomics (bandwidth bound)
//  - ODE RK4 fully fused: weights f16 in LDS, z-state f32, readout fused
// ---------------------------------------------------------------------------

#define N_NODES 50000
#define N_EDGES 800000
#define ETOT    (N_EDGES + N_NODES)
#define F_IN    12
#define HID     64
#define HEADS   4
#define TSTEPS  12
#define HID2    128
#define NEG_SLOPE 0.2f

typedef __attribute__((ext_vector_type(16))) _Float16 v16h;
typedef __attribute__((ext_vector_type(8)))  float    v8f;

__device__ __forceinline__ v8f wmma_f16(v16h a, v16h b, v8f c) {
  // (neg_a, A, neg_b, B, c_mod, C, reuse_a, reuse_b)
  return __builtin_amdgcn_wmma_f32_16x16x32_f16(false, a, false, b, (short)0, c,
                                                false, false);
}

// Fragment-slot maps (half index within a 32-lane x 16-half fragment buffer).
// A tile (16x32 f16): lane = row + 16*k[3]; vgpr-pair v = 4*k[4] + k[2:1]
__device__ __forceinline__ int a_slot(int row, int k) {
  int lane = row + (((k >> 3) & 1) << 4);
  int v = ((k >> 4) << 2) + ((k & 7) >> 1);
  return lane * 16 + 2 * v + (k & 1);
}
// B tile (32x16 f16): lane = col + 16*k[3]; j = 8*k[4] + k[2:0]
__device__ __forceinline__ int b_slot(int k, int col) {
  int lane = col + (((k >> 3) & 1) << 4);
  return lane * 16 + (((k >> 4) & 1) << 3) + (k & 7);
}

__device__ __forceinline__ float leaky(float x) {
  return x > 0.0f ? x : NEG_SLOPE * x;
}

__device__ __forceinline__ void atomic_max_f32(float* addr, float val) {
  unsigned int* ua = (unsigned int*)addr;
  unsigned int old = *ua;
  while (__uint_as_float(old) < val) {
    unsigned int assumed = old;
    old = atomicCAS(ua, assumed, __float_as_uint(val));
    if (old == assumed) break;
  }
}

// ---------------------------------------------------------------------------
// One-time: pack f32 weight [K, ncols] into fragment-major f16 image,
// zero-padded in K to KP. frag = (k/32)*(ncols/16) + (c/16).
// ---------------------------------------------------------------------------
__global__ void pack_b_kernel(const float* __restrict__ W,
                              _Float16* __restrict__ P,
                              int K, int KP, int ncols) {
  int idx = blockIdx.x * blockDim.x + threadIdx.x;
  if (idx >= KP * ncols) return;
  int k = idx / ncols, c = idx % ncols;
  float v = (k < K) ? W[k * ncols + c] : 0.0f;
  int frag = (k >> 5) * (ncols >> 4) + (c >> 4);
  P[frag * 512 + b_slot(k & 31, c & 15)] = (_Float16)v;
}

// ---------------------------------------------------------------------------
// xw = X[N,K] @ W[K,256]   (WP: pre-packed fragment-major f16; KP in {32,64})
// block: 128 threads = 4 waves; block does 16 rows x 256 cols.
// ---------------------------------------------------------------------------
__global__ __launch_bounds__(128) void proj_kernel(
    const float* __restrict__ X, const _Float16* __restrict__ WP,
    float* __restrict__ XW, int K, int KP) {
  __shared__ v16h wlB[2 * 16 * 32];   // 32 KB
  __shared__ v16h xlA[2 * 32];        //  2 KB
  int tid  = threadIdx.x;
  int row0 = blockIdx.x * 16;         // N % 16 == 0
  int nk = KP >> 5;
  const v16h* wsrc = (const v16h*)WP;
  for (int i = tid; i < nk * 16 * 32; i += 128) wlB[i] = wsrc[i];
  _Float16* xlh = (_Float16*)xlA;
  for (int idx = tid; idx < 16 * KP; idx += 128) {
    int r = idx / KP, k = idx % KP;
    float v = (k < K) ? X[(size_t)(row0 + r) * K + k] : 0.0f;
    xlh[((k >> 5) << 9) + a_slot(r, k & 31)] = (_Float16)v;
  }
  __syncthreads();
  int wave = tid >> 5, lane = tid & 31;
  int hi = (lane >> 4) & 1, cl = lane & 15;
  v16h a0 = xlA[lane];
  v16h a1 = a0;
  if (nk > 1) a1 = xlA[32 + lane];
#pragma unroll
  for (int nt = 0; nt < 4; ++nt) {
    int nti = wave * 4 + nt;
    v8f acc = {};
    acc = wmma_f16(a0, wlB[nti * 32 + lane], acc);
    if (nk > 1) acc = wmma_f16(a1, wlB[(16 + nti) * 32 + lane], acc);
#pragma unroll
    for (int r = 0; r < 8; ++r)
      XW[(size_t)(row0 + r + 8 * hi) * 256 + nti * 16 + cl] = acc[r];
  }
}

// ---------------------------------------------------------------------------
// per (node, head): attention logits, init segment-max/denom, zero ACC
// ---------------------------------------------------------------------------
__global__ void al_init_kernel(const float* __restrict__ XW,
                               const float* __restrict__ a_src,
                               const float* __restrict__ a_dst,
                               float* ALS, float* ALD, float* EMAX, float* DEN,
                               float* ACC) {
  int idx = blockIdx.x * blockDim.x + threadIdx.x;
  if (idx >= N_NODES * HEADS) return;
  int n = idx >> 2, h = idx & 3;
  const float* xw = XW + (size_t)n * 256 + h * 64;
  const float* as = a_src + h * 64;
  const float* ad = a_dst + h * 64;
  float ss = 0.0f, sd = 0.0f;
  for (int c = 0; c < 64; ++c) { float v = xw[c]; ss += v * as[c]; sd += v * ad[c]; }
  ALS[idx] = ss;
  ALD[idx] = sd;
  EMAX[idx] = -3.402823e38f;
  DEN[idx]  = 0.0f;
  float* acc = ACC + (size_t)n * 256 + h * 64;
  for (int c = 0; c < 64; ++c) acc[c] = 0.0f;
}

// ---------------------------------------------------------------------------
// segment max over dst (edges + self-loops)
// ---------------------------------------------------------------------------
__global__ void edge_max_kernel(const long long* __restrict__ ei,
                                const float* __restrict__ ALS,
                                const float* __restrict__ ALD,
                                float* __restrict__ EMAX) {
  int i = blockIdx.x * blockDim.x + threadIdx.x;
  if (i >= ETOT) return;
  int s, d;
  if (i < N_EDGES) { s = (int)ei[i]; d = (int)ei[N_EDGES + i]; }
  else             { s = d = i - N_EDGES; }
#pragma unroll
  for (int h = 0; h < 4; ++h) {
    float e = leaky(ALS[s * 4 + h] + ALD[d * 4 + h]);
    atomic_max_f32(&EMAX[d * 4 + h], e);
  }
}

// ---------------------------------------------------------------------------
// segment sum of exp(e - max)
// ---------------------------------------------------------------------------
__global__ void edge_den_kernel(const long long* __restrict__ ei,
                                const float* __restrict__ ALS,
                                const float* __restrict__ ALD,
                                const float* __restrict__ EMAX,
                                float* __restrict__ DEN) {
  int i = blockIdx.x * blockDim.x + threadIdx.x;
  if (i >= ETOT) return;
  int s, d;
  if (i < N_EDGES) { s = (int)ei[i]; d = (int)ei[N_EDGES + i]; }
  else             { s = d = i - N_EDGES; }
#pragma unroll
  for (int h = 0; h < 4; ++h) {
    float e = leaky(ALS[s * 4 + h] + ALD[d * 4 + h]);
    atomicAdd(&DEN[d * 4 + h], __expf(e - EMAX[d * 4 + h]));
  }
}

// ---------------------------------------------------------------------------
// message pass: one wave per edge; 256 (head,channel) values per edge,
// 8 per lane -> atomic f32 adds into ACC[dst]
// ---------------------------------------------------------------------------
__global__ void edge_msg_kernel(const long long* __restrict__ ei,
                                const float* __restrict__ ALS,
                                const float* __restrict__ ALD,
                                const float* __restrict__ EMAX,
                                const float* __restrict__ DEN,
                                const float* __restrict__ XW,
                                float* __restrict__ ACC) {
  int g = blockIdx.x * blockDim.x + threadIdx.x;
  int e = g >> 5, lane = g & 31;
  if (e >= ETOT) return;
  int s, d;
  if (e < N_EDGES) { s = (int)ei[e]; d = (int)ei[N_EDGES + e]; }
  else             { s = d = e - N_EDGES; }
  const float* xw  = XW  + (size_t)s * 256;
  float*       acc = ACC + (size_t)d * 256;
  __builtin_prefetch(xw, 0, 0);   // global_prefetch_b8
#pragma unroll
  for (int rep = 0; rep < 8; ++rep) {
    int idx = rep * 32 + lane;
    int h = idx >> 6;
    float ev = leaky(ALS[s * 4 + h] + ALD[d * 4 + h]);
    float alpha = __expf(ev - EMAX[d * 4 + h]) / DEN[d * 4 + h];
    atomicAdd(&acc[idx], xw[idx] * alpha);
  }
}

// ---------------------------------------------------------------------------
// head-mean + bias + relu
// ---------------------------------------------------------------------------
__global__ void finalize_kernel(const float* __restrict__ ACC,
                                const float* __restrict__ bias,
                                float* __restrict__ Z) {
  int idx = blockIdx.x * blockDim.x + threadIdx.x;
  if (idx >= N_NODES * HID) return;
  int n = idx >> 6, c = idx & 63;
  const float* a = ACC + (size_t)n * 256 + c;
  float v = 0.25f * (a[0] + a[64] + a[128] + a[192]) + bias[c];
  Z[idx] = v > 0.0f ? v : 0.0f;
}

// ---------------------------------------------------------------------------
// Fused RK4 ODE + readout. 64 threads = 2 waves, each wave owns 16 rows.
// w1B: [kt(2)][nt(8)] frags; w2B: [kt(4)][nt(4)] frags (pre-packed f16).
// zinA: [kt(2)] A-frags; hhA: [kt(4)] A-frags. zb: f32 carried state.
// ---------------------------------------------------------------------------
__global__ __launch_bounds__(64) void ode_kernel(
    const float* __restrict__ Z,
    const _Float16* __restrict__ w1p, const float* __restrict__ b_o1,
    const _Float16* __restrict__ w2p, const float* __restrict__ b_o2,
    const float* __restrict__ wf,  const float* __restrict__ bf,
    const float* __restrict__ times, float* __restrict__ out) {
  __shared__ v16h w1B[16 * 32];             // 16 KB
  __shared__ v16h w2B[16 * 32];             // 16 KB
  __shared__ float b1l[HID2], b2l[HID], wfl[HID];
  __shared__ float zb[2][16 * HID];         //  8 KB f32 carried state
  __shared__ v16h zinA[2][2 * 32];          //  4 KB f16 eval input frags
  __shared__ v16h hhA[2][4 * 32];           //  8 KB f16 hidden frags
  int tid = threadIdx.x;
  const v16h* s1 = (const v16h*)w1p;
  const v16h* s2 = (const v16h*)w2p;
  for (int i = tid; i < 512; i += 64) { w1B[i] = s1[i]; w2B[i] = s2[i]; }
  for (int i = tid; i < HID2; i += 64) b1l[i] = b_o1[i];
  for (int i = tid; i < HID; i += 64) { b2l[i] = b_o2[i]; wfl[i] = wf[i]; }
  __syncthreads();

  int wave = tid >> 5, lane = tid & 31;
  int row0 = blockIdx.x * 32 + wave * 16;
  if (row0 >= N_NODES) return;              // wave-uniform exit: EXEC stays full
  float*    zbw = zb[wave];
  _Float16* zih = (_Float16*)zinA[wave];
  _Float16* hhh = (_Float16*)hhA[wave];

  for (int i = lane; i < 16 * HID; i += 32)
    zbw[i] = Z[(size_t)row0 * HID + i];
  asm volatile("s_wait_dscnt 0x0" ::: "memory");

  float bfv = bf[0];
  int hi = (lane >> 4) & 1, cl = lane & 15;

  auto predict = [&](int tp) {              // z . wf + bf  (lanes 0..15 = rows)
    if (lane < 16) {
      float p = bfv;
      for (int c = 0; c < HID; ++c) p += zbw[lane * HID + c] * wfl[c];
      out[(size_t)(row0 + lane) * TSTEPS + tp] = p;
    }
  };
  predict(0);

  const float stage_c[3] = {0.5f, 0.5f, 1.0f};
  const float stage_w[4] = {1.0f, 2.0f, 2.0f, 1.0f};
  v8f vzero = {};

  for (int t = 1; t < TSTEPS; ++t) {
    float dt = times[t] - times[t - 1];
    // zin = zb (scatter straight into A-fragment slots)
    for (int i = lane; i < 16 * HID; i += 32) {
      int row = i >> 6, k = i & 63;
      zih[((k >> 5) << 9) + a_slot(row, k & 31)] = (_Float16)zbw[i];
    }
    asm volatile("s_wait_dscnt 0x0" ::: "memory");

    v8f acc[4];
#pragma unroll
    for (int n1 = 0; n1 < 4; ++n1) acc[n1] = vzero;

    for (int s = 0; s < 4; ++s) {
      // ---- h = tanh(zin @ w_o1 + b_o1): A-frags hoisted, 2 WMMA per tile
      v16h a0 = zinA[wave][lane];
      v16h a1 = zinA[wave][32 + lane];
#pragma unroll
      for (int n2 = 0; n2 < 8; ++n2) {
        v8f c = vzero;
        c = wmma_f16(a0, w1B[n2 * 32 + lane], c);
        c = wmma_f16(a1, w1B[(8 + n2) * 32 + lane], c);
        float bias = b1l[n2 * 16 + cl];
        int col = n2 * 16 + cl;             // k-dim of GEMM2
#pragma unroll
        for (int r = 0; r < 8; ++r)
          hhh[((col >> 5) << 9) + a_slot(r + 8 * hi, col & 31)] =
              (_Float16)tanhf(c[r] + bias);
      }
      asm volatile("s_wait_dscnt 0x0" ::: "memory");

      // ---- f = h @ w_o2 + b_o2: A-frags hoisted, 4 WMMA per tile; fold RK4
      v16h ha[4];
#pragma unroll
      for (int kt = 0; kt < 4; ++kt) ha[kt] = hhA[wave][kt * 32 + lane];
#pragma unroll
      for (int n1 = 0; n1 < 4; ++n1) {
        v8f c2 = vzero;
#pragma unroll
        for (int kt = 0; kt < 4; ++kt)
          c2 = wmma_f16(ha[kt], w2B[(kt * 4 + n1) * 32 + lane], c2);
        float bias2 = b2l[n1 * 16 + cl];
        int col = n1 * 16 + cl;             // k-dim of zin
#pragma unroll
        for (int r = 0; r < 8; ++r) {
          float f = c2[r] + bias2;
          acc[n1][r] += stage_w[s] * f;
          if (s < 3) {
            int row = r + 8 * hi;
            zih[((col >> 5) << 9) + a_slot(row, col & 31)] =
                (_Float16)(zbw[row * HID + col] + stage_c[s] * dt * f);
          }
        }
      }
      asm volatile("s_wait_dscnt 0x0" ::: "memory");
    }

    float sc = dt * (1.0f / 6.0f);
#pragma unroll
    for (int n1 = 0; n1 < 4; ++n1)
#pragma unroll
      for (int r = 0; r < 8; ++r)
        zbw[(r + 8 * hi) * HID + n1 * 16 + cl] += sc * acc[n1][r];
    asm volatile("s_wait_dscnt 0x0" ::: "memory");
    predict(t);
  }
}

// ---------------------------------------------------------------------------
extern "C" void kernel_launch(void* const* d_in, const int* in_sizes, int n_in,
                              void* d_out, int out_size, void* d_ws, size_t ws_size,
                              hipStream_t stream) {
  (void)in_sizes; (void)n_in; (void)out_size; (void)ws_size;
  const float*     x     = (const float*)d_in[0];
  const long long* ei    = (const long long*)d_in[1];   // int64 edge_index [2,E]
  const float*     times = (const float*)d_in[3];
  const float* W1  = (const float*)d_in[4];
  const float* as1 = (const float*)d_in[5];
  const float* ad1 = (const float*)d_in[6];
  const float* bg1 = (const float*)d_in[7];
  const float* W2  = (const float*)d_in[8];
  const float* as2 = (const float*)d_in[9];
  const float* ad2 = (const float*)d_in[10];
  const float* bg2 = (const float*)d_in[11];
  const float* wo1 = (const float*)d_in[12];
  const float* bo1 = (const float*)d_in[13];
  const float* wo2 = (const float*)d_in[14];
  const float* bo2 = (const float*)d_in[15];
  const float* wfp = (const float*)d_in[16];
  const float* bfp = (const float*)d_in[17];
  float* out = (float*)d_out;

  // workspace layout (floats, then f16 packed-weight images; 32B aligned)
  float* XW  = (float*)d_ws;                         // N*256
  float* ACC = XW  + (size_t)N_NODES * 256;          // N*256
  float* ALS = ACC + (size_t)N_NODES * 256;          // N*4
  float* ALD = ALS + (size_t)N_NODES * 4;            // N*4
  float* EMX = ALD + (size_t)N_NODES * 4;            // N*4
  float* DEN = EMX + (size_t)N_NODES * 4;            // N*4
  float* Z   = DEN + (size_t)N_NODES * 4;            // N*64
  _Float16* WP1 = (_Float16*)(Z + (size_t)N_NODES * 64);  // 32*256
  _Float16* WP2 = WP1 + 32 * 256;                          // 64*256
  _Float16* WO1 = WP2 + 64 * 256;                          // 64*128
  _Float16* WO2 = WO1 + 64 * 128;                          // 128*64

  auto cdiv = [](long long a, long long b) { return (int)((a + b - 1) / b); };

  // one-time fragment-major weight packing (tiny)
  pack_b_kernel<<<cdiv(32 * 256, 256), 256, 0, stream>>>(W1, WP1, F_IN, 32, 256);
  pack_b_kernel<<<cdiv(64 * 256, 256), 256, 0, stream>>>(W2, WP2, HID, 64, 256);
  pack_b_kernel<<<cdiv(64 * 128, 256), 256, 0, stream>>>(wo1, WO1, HID, 64, HID2);
  pack_b_kernel<<<cdiv(128 * 64, 256), 256, 0, stream>>>(wo2, WO2, HID2, 128, HID);

  auto layer = [&](const float* X, int K, int KP, const _Float16* WP,
                   const float* asrc, const float* adst, const float* bias) {
    proj_kernel<<<N_NODES / 16, 128, 0, stream>>>(X, WP, XW, K, KP);
    al_init_kernel<<<cdiv((long long)N_NODES * 4, 256), 256, 0, stream>>>(
        XW, asrc, adst, ALS, ALD, EMX, DEN, ACC);
    edge_max_kernel<<<cdiv(ETOT, 256), 256, 0, stream>>>(ei, ALS, ALD, EMX);
    edge_den_kernel<<<cdiv(ETOT, 256), 256, 0, stream>>>(ei, ALS, ALD, EMX, DEN);
    edge_msg_kernel<<<cdiv((long long)ETOT * 32, 256), 256, 0, stream>>>(
        ei, ALS, ALD, EMX, DEN, XW, ACC);
    finalize_kernel<<<cdiv((long long)N_NODES * HID, 256), 256, 0, stream>>>(
        ACC, bias, Z);
  };

  layer(x, F_IN, 32, WP1, as1, ad1, bg1);  // GAT layer 1 (K=12 padded to 32)
  layer(Z, HID, 64, WP2, as2, ad2, bg2);   // GAT layer 2 (K=64)

  ode_kernel<<<cdiv(N_NODES, 32), 64, 0, stream>>>(
      Z, WO1, bo1, WO2, bo2, wfp, bfp, times, out);
}
